// MaskedAttention_15814069584503
// MI455X (gfx1250) — matrix operand order repaired and checked by reading, not compiled, not run on hardware
//
#include <hip/hip_runtime.h>
#include <hip/hip_bf16.h>

#define B_      2
#define N_      2048
#define C_      768
#define H_      12
#define DH_     64
#define QKV_LD  2304            // 3*C_
#define SCALE_  0.125f          // DH^-0.5
#define MASKV  (-3.402823466e38f)

typedef __attribute__((ext_vector_type(16))) __bf16          v16bf;
typedef __attribute__((ext_vector_type(8)))  float           v8f;
typedef __attribute__((ext_vector_type(8)))  unsigned short  v8us;
typedef __attribute__((ext_vector_type(4)))  unsigned int    u32x4;
typedef __attribute__((ext_vector_type(8)))  int             i32x8;
typedef __attribute__((ext_vector_type(4)))  int             i32x4;

#if defined(__has_builtin)
#  if __has_builtin(__builtin_amdgcn_tensor_load_to_lds) && \
      __has_builtin(__builtin_amdgcn_s_wait_tensorcnt)
#    define HAVE_TDM 1
#  endif
#endif
#ifndef HAVE_TDM
#  define HAVE_TDM 0
#endif

union FragBF {
  v16bf v;
  v8us  h[2];
  unsigned short u[16];
};

__device__ __forceinline__ unsigned short f2bf(float f) {
  unsigned int u = __float_as_uint(f);
  u += 0x7fffu + ((u >> 16) & 1u);          // round-to-nearest-even
  return (unsigned short)(u >> 16);
}

// A-matrix 16x32 bf16 fragment: lanes 0-15 row=lane, K 0..15;
// lanes 16-31 row=lane-16, K 16..31. Memory row-major, K contiguous.
__device__ __forceinline__ v16bf load_frag_a(const unsigned short* p, int row, int ld, int k0) {
  const int lane = threadIdx.x & 31;
  const unsigned short* base = p + (size_t)(row + (lane & 15)) * ld + k0 + ((lane >> 4) << 4);
  FragBF f;
  f.h[0] = *(const v8us*)(base);
  f.h[1] = *(const v8us*)(base + 8);
  return f.v;
}

// B-matrix 32x16 bf16 fragment: lanes 0-15 col=lane, K 0..15; lanes 16-31 col=lane-16, K 16..31.
// Column n of B is contiguous-K at p + n*ld (i.e. Bm[n][k] row-major, B = Bm^T).
__device__ __forceinline__ v16bf load_frag_b(const unsigned short* p, int col, int ld, int k0) {
  const int lane = threadIdx.x & 31;
  const unsigned short* base = p + (size_t)(col + (lane & 15)) * ld + k0 + ((lane >> 4) << 4);
  FragBF f;
  f.h[0] = *(const v8us*)(base);
  f.h[1] = *(const v8us*)(base + 8);
  return f.v;
}

__device__ __forceinline__ v8f wmma_bf16(v16bf a, v16bf b, v8f c) {
  return __builtin_amdgcn_wmma_f32_16x16x32_bf16(false, a, false, b, (short)0, c, false, false);
}

// ---- 16-lane butterfly reductions in pure VALU via DPP16 (no LDS, no dscnt waits).
template <int CTRL>
__device__ __forceinline__ float dpp_swap(float v) {
  return __int_as_float(
      __builtin_amdgcn_update_dpp(0, __float_as_int(v), CTRL, 0xF, 0xF, true));
}
__device__ __forceinline__ float red_max16(float v) {
  v = fmaxf(v, dpp_swap<0xB1>(v));   // quad_perm [1,0,3,2]
  v = fmaxf(v, dpp_swap<0x4E>(v));   // quad_perm [2,3,0,1]
  v = fmaxf(v, dpp_swap<0x141>(v));  // row_half_mirror
  v = fmaxf(v, dpp_swap<0x140>(v));  // row_mirror
  return v;
}
__device__ __forceinline__ float red_sum16(float v) {
  v += dpp_swap<0xB1>(v);
  v += dpp_swap<0x4E>(v);
  v += dpp_swap<0x141>(v);
  v += dpp_swap<0x140>(v);
  return v;
}

__global__ void cvt_f32_bf16(const float* __restrict__ in, unsigned short* __restrict__ out, int n) {
  int i = blockIdx.x * blockDim.x + threadIdx.x;
  if (i < n) out[i] = f2bf(in[i]);
}

// C[M,Nn] = A[M,K](bf16) * Bm[Nn,K]^T(bf16), one wave -> 16x64 tile, 4 f32 accumulators.
template <bool OUT_F32>
__global__ void gemm_bf16_wmma(const unsigned short* __restrict__ A,
                               const unsigned short* __restrict__ Bm,
                               const float* __restrict__ bias,
                               void* __restrict__ Cout,
                               int M, int Nn, int K) {
  const int wave = blockIdx.x * (blockDim.x >> 5) + (threadIdx.x >> 5);
  const int ntiles = Nn >> 6;
  const int m0 = (wave / ntiles) << 4;
  const int n0 = (wave % ntiles) << 6;
  if (m0 >= M) return;
  const int lane = threadIdx.x & 31;

  v8f acc[4];
#pragma unroll
  for (int j = 0; j < 4; ++j)
#pragma unroll
    for (int e = 0; e < 8; ++e) acc[j][e] = 0.0f;

  for (int k0 = 0; k0 < K; k0 += 32) {
    v16bf a = load_frag_a(A, m0, K, k0);
#pragma unroll
    for (int j = 0; j < 4; ++j) {
      v16bf b = load_frag_b(Bm, n0 + (j << 4), K, k0);
      acc[j] = wmma_bf16(a, b, acc[j]);
    }
  }

  const int rowBase = m0 + ((lane >> 4) << 3);
  const int colBase = n0 + (lane & 15);
#pragma unroll
  for (int j = 0; j < 4; ++j) {
    const int c = colBase + (j << 4);
#pragma unroll
    for (int e = 0; e < 8; ++e) {
      const int r = rowBase + e;
      if constexpr (OUT_F32) {
        ((float*)Cout)[(size_t)r * Nn + c] = acc[j][e] + bias[c];
      } else {
        ((unsigned short*)Cout)[(size_t)r * Nn + c] = f2bf(acc[j][e]);
      }
    }
  }
}

#if HAVE_TDM
// TDM 2D tile load: 32 rows x 64 bf16, row pitch QKV_LD elements -> LDS.
// D# packing per CDNA5 ISA sec 8.3 / 8.4; 6-arg builtin (clang-23 / therock-10.0).
__device__ __forceinline__ void tdm_load_tile_32x64(unsigned lds_byte_addr, const void* gptr) {
  const unsigned long long ga = (unsigned long long)(size_t)gptr;
  u32x4 g0;
  g0[0] = 1u;                                               // count=1, no gather
  g0[1] = lds_byte_addr;                                    // lds_addr
  g0[2] = (unsigned)(ga & 0xFFFFFFFFull);                   // global_addr[31:0]
  g0[3] = (unsigned)((ga >> 32) & 0x01FFFFFFull)            // global_addr[56:32]
        | (2u << 30);                                       // type = 2 (image)
  i32x8 g1;
  g1[0] = 0x00010000;                                       // data_size = 1 -> 2 bytes
  g1[1] = (int)(64u << 16);                                 // tensor_dim0 = 64 (lo16)
  g1[2] = (int)((unsigned)N_ << 16);                        // tensor_dim1 = 2048 (lo16)
  g1[3] = (int)(64u << 16);                                 // tile_dim0 = 64
  g1[4] = 32;                                               // tile_dim1 = 32, tile_dim2 = 0
  g1[5] = QKV_LD;                                           // tensor_dim0_stride lo32
  g1[6] = 0;                                                // stride hi / dim1_stride
  g1[7] = 0;
  i32x4 z4;
  z4[0] = 0; z4[1] = 0; z4[2] = 0; z4[3] = 0;
  i32x8 z8;
#pragma unroll
  for (int i = 0; i < 8; ++i) z8[i] = 0;
  __builtin_amdgcn_tensor_load_to_lds(g0, g1, z4, z4, z8, 0);
}
#endif

__device__ __forceinline__ void coop_load_tile_32x64(unsigned short* dst,
                                                     const unsigned short* src,
                                                     int tid, int nthr) {
  for (int i = tid; i < 32 * 8; i += nthr) {
    const int r = i >> 3, c = (i & 7) << 3;
    *(v8us*)(dst + r * 64 + c) = *(const v8us*)(src + (size_t)r * QKV_LD + c);
  }
}

// Flash attention: grid (B*H, N/64), 128 threads (4 waves), each wave = 16 query rows.
// K/V 32x64 bf16 tiles staged in LDS (TDM when available), shared by all 4 waves.
__global__ void attn_flash_wmma(const unsigned short* __restrict__ qkv,
                                const int* __restrict__ mask,
                                unsigned short* __restrict__ Obuf) {
  const int b    = blockIdx.x / H_;
  const int h    = blockIdx.x % H_;
  const int wid  = threadIdx.x >> 5;
  const int lane = threadIdx.x & 31;
  const int q0   = blockIdx.y * 64 + wid * 16;

  __shared__ unsigned short kT[32 * 64];        // K tile: [key][dh]
  __shared__ unsigned short vT[32 * 64];        // V tile: [key][dh]
  __shared__ unsigned short pLDS[4][16 * 32];   // per-wave P bounce
  unsigned short* myP = pLDS[wid];

  const size_t rowB = (size_t)b * N_;
  const unsigned short* qptr = qkv + rowB * QKV_LD + h * DH_;
  const unsigned short* kptr = qptr + C_;
  const unsigned short* vptr = qptr + 2 * C_;
  const int* mrow = mask + b * N_;

  const v16bf aQ0 = load_frag_a(qptr, q0, QKV_LD, 0);
  const v16bf aQ1 = load_frag_a(qptr, q0, QKV_LD, 32);

  bool rm[8];
#pragma unroll
  for (int e = 0; e < 8; ++e)
    rm[e] = mrow[q0 + ((lane >> 4) << 3) + e] != 0;

  // Hoist the entire column mask into two per-lane 64-bit registers:
  // bit i of cmLo = mask[i*32 + (lane&15)], bit i of cmHi = mask[i*32+16+(lane&15)].
  // Removes per-iteration global loads + s_wait_loadcnt 0 from the hot loop.
  unsigned long long cmLo = 0ull, cmHi = 0ull;
  {
    const int l16 = lane & 15;
    for (int i = 0; i < N_ / 32; ++i) {
      cmLo |= (unsigned long long)(mrow[i * 32 + l16]      != 0) << i;
      cmHi |= (unsigned long long)(mrow[i * 32 + 16 + l16] != 0) << i;
    }
  }

  float m_run[8], l_run[8];
  v8f o[4];
#pragma unroll
  for (int e = 0; e < 8; ++e) { m_run[e] = MASKV; l_run[e] = 0.0f; }
#pragma unroll
  for (int j = 0; j < 4; ++j)
#pragma unroll
    for (int e = 0; e < 8; ++e) o[j][e] = 0.0f;

  for (int kb = 0; kb < N_; kb += 32) {
    __syncthreads();                            // all waves done with previous tiles
#if HAVE_TDM
    if (wid == 0) {
      tdm_load_tile_32x64((unsigned)(size_t)(void*)kT, kptr + (size_t)kb * QKV_LD);
      tdm_load_tile_32x64((unsigned)(size_t)(void*)vT, vptr + (size_t)kb * QKV_LD);
      __builtin_amdgcn_s_wait_tensorcnt(0);
    }
#else
    coop_load_tile_32x64(kT, kptr + (size_t)kb * QKV_LD, threadIdx.x, blockDim.x);
    coop_load_tile_32x64(vT, vptr + (size_t)kb * QKV_LD, threadIdx.x, blockDim.x);
#endif
    __syncthreads();                            // tiles visible to all waves

    v8f s0, s1;
#pragma unroll
    for (int e = 0; e < 8; ++e) { s0[e] = 0.0f; s1[e] = 0.0f; }

    {   // S = Q * K^T  (two 16x16 key tiles, Dh=64 -> two k-steps each), K from LDS
      v16bf bk;
      bk = load_frag_b(kT, 0,  64, 0);  s0 = wmma_bf16(aQ0, bk, s0);
      bk = load_frag_b(kT, 0,  64, 32); s0 = wmma_bf16(aQ1, bk, s0);
      bk = load_frag_b(kT, 16, 64, 0);  s1 = wmma_bf16(aQ0, bk, s1);
      bk = load_frag_b(kT, 16, 64, 32); s1 = wmma_bf16(aQ1, bk, s1);
    }

    const int it  = kb >> 5;
    const bool cm0 = (cmLo >> it) & 1ull;
    const bool cm1 = (cmHi >> it) & 1ull;

    float alpha[8];
#pragma unroll
    for (int e = 0; e < 8; ++e) {
      const float v0 = (rm[e] && cm0) ? s0[e] * SCALE_ : MASKV;
      const float v1 = (rm[e] && cm1) ? s1[e] * SCALE_ : MASKV;
      const float tmax = red_max16(fmaxf(v0, v1));        // DPP-only butterfly
      const float mnew = fmaxf(m_run[e], tmax);
      const float a    = __expf(m_run[e] - mnew);
      const float p0   = __expf(v0 - mnew);
      const float p1   = __expf(v1 - mnew);
      l_run[e] = l_run[e] * a + red_sum16(p0 + p1);
      m_run[e] = mnew;
      alpha[e] = a;
      s0[e] = p0; s1[e] = p1;
    }
#pragma unroll
    for (int j = 0; j < 4; ++j)
#pragma unroll
      for (int e = 0; e < 8; ++e) o[j][e] *= alpha[e];

    // P (C-layout) -> LDS -> A-layout fragment for P*V (wave-private region)
#pragma unroll
    for (int e = 0; e < 8; ++e) {
      const int r = ((lane >> 4) << 3) + e;
      myP[r * 32 + (lane & 15)]      = f2bf(s0[e]);
      myP[r * 32 + 16 + (lane & 15)] = f2bf(s1[e]);
    }
    asm volatile("s_wait_dscnt 0x0" ::: "memory");
    __builtin_amdgcn_wave_barrier();

    FragBF pf;
    {
      const unsigned short* pb = myP + (lane & 15) * 32 + ((lane >> 4) << 4);
      pf.h[0] = *(const v8us*)pb;
      pf.h[1] = *(const v8us*)(pb + 8);
    }

    // O += P * V   (V B-fragment gathered from LDS tile: col = head dim, K = key)
    const int kkBase = (lane >> 4) << 4;
#pragma unroll
    for (int j = 0; j < 4; ++j) {
      FragBF vf;
      const int cbase = (j << 4) + (lane & 15);
#pragma unroll
      for (int e = 0; e < 16; ++e)
        vf.u[e] = vT[(kkBase + e) * 64 + cbase];
      o[j] = wmma_bf16(pf.v, vf.v, o[j]);
    }
  }

#pragma unroll
  for (int e = 0; e < 8; ++e) l_run[e] = 1.0f / l_run[e];
  const int rowBase = q0 + ((lane >> 4) << 3);
#pragma unroll
  for (int j = 0; j < 4; ++j) {
    const int c = h * DH_ + (j << 4) + (lane & 15);
#pragma unroll
    for (int e = 0; e < 8; ++e)
      Obuf[(rowB + rowBase + e) * (size_t)C_ + c] = f2bf(o[j][e] * l_run[e]);
  }
}

extern "C" void kernel_launch(void* const* d_in, const int* in_sizes, int n_in,
                              void* d_out, int out_size, void* d_ws, size_t ws_size,
                              hipStream_t stream) {
  const float* x      = (const float*)d_in[0];   // (B,N,1,C)
  const int*   mask   = (const int*)  d_in[1];   // (B,N)
  const float* w_qkv  = (const float*)d_in[2];   // (3C,C)
  const float* w_proj = (const float*)d_in[3];   // (C,C)
  const float* b_proj = (const float*)d_in[4];   // (C,)
  float* out = (float*)d_out;                    // (B,N,C)

  const int M = B_ * N_;                         // 4096 rows
  char* ws = (char*)d_ws;
  size_t off = 0;
  auto alloc = [&](size_t bytes) -> char* {
    char* p = ws + off;
    off = (off + bytes + 255) & ~(size_t)255;
    return p;
  };
  unsigned short* xb     = (unsigned short*)alloc((size_t)M * C_ * 2);
  unsigned short* wqkvb  = (unsigned short*)alloc((size_t)QKV_LD * C_ * 2);
  unsigned short* wprojb = (unsigned short*)alloc((size_t)C_ * C_ * 2);
  unsigned short* qkvb   = (unsigned short*)alloc((size_t)M * QKV_LD * 2);
  unsigned short* ob     = (unsigned short*)alloc((size_t)M * C_ * 2);
  (void)ws_size; (void)in_sizes; (void)n_in; (void)out_size;

  // 1. convert inputs to bf16
  {
    int n1 = M * C_, n2 = QKV_LD * C_, n3 = C_ * C_;
    cvt_f32_bf16<<<(n1 + 255) / 256, 256, 0, stream>>>(x, xb, n1);
    cvt_f32_bf16<<<(n2 + 255) / 256, 256, 0, stream>>>(w_qkv, wqkvb, n2);
    cvt_f32_bf16<<<(n3 + 255) / 256, 256, 0, stream>>>(w_proj, wprojb, n3);
  }

  // 2. fused QKV projection: qkv[M, 3C] = x @ w_qkv^T
  {
    int waves = (M / 16) * (QKV_LD / 64);        // 9216
    gemm_bf16_wmma<false><<<(waves + 7) / 8, 256, 0, stream>>>(
        xb, wqkvb, nullptr, qkvb, M, QKV_LD, C_);
  }

  // 3. flash attention -> O in (B, N, H*Dh) = (B, N, C) bf16
  attn_flash_wmma<<<dim3(B_ * H_, N_ / 64), 128, 0, stream>>>(qkvb, mask, ob);

  // 4. output projection: out = O @ w_proj^T + b_proj  (f32)
  {
    int waves = (M / 16) * (C_ / 64);            // 3072
    gemm_bf16_wmma<true><<<(waves + 7) / 8, 256, 0, stream>>>(
        ob, wprojb, b_proj, out, M, C_, C_);
  }
}
